// TransformerEncoder_59889023975699
// MI455X (gfx1250) — matrix operand order repaired and checked by reading, not compiled, not run on hardware
//
#include <hip/hip_runtime.h>
#include <hip/hip_bf16.h>
#include <math.h>

// ---------------------------------------------------------------------------
// Types for WMMA (CDNA5 gfx1250, wave32)
// ---------------------------------------------------------------------------
typedef __attribute__((ext_vector_type(16))) _Float16 v16h;
typedef __attribute__((ext_vector_type(8)))  _Float16 v8h;
typedef __attribute__((ext_vector_type(8)))  float    v8f;

__device__ __forceinline__ v8f wmma_f16(v16h a, v16h b, v8f c) {
  // D = A(16x32,f16) * B(32x16,f16) + C(16x16,f32)
  return __builtin_amdgcn_wmma_f32_16x16x32_f16(
      /*neg_a=*/false, a, /*neg_b=*/false, b,
      /*c_mod=*/(short)0, c, /*reuse_a=*/false, /*reuse_b=*/false);
}

__device__ __forceinline__ v16h cat8(v8h a, v8h b) {
  v16h r;
#pragma unroll
  for (int i = 0; i < 8; ++i) { r[i] = a[i]; r[i + 8] = b[i]; }
  return r;
}

// A-matrix fragment (16x32 f16): lane&15 -> row M, lanes 0-15 K=0-15,
// lanes 16-31 K=16-31 (contiguous 16 halves per half-wave).
__device__ __forceinline__ v16h frag_a(const _Float16* base, int rowstride, int lane) {
  const int ln = lane & 15, hl = lane >> 4;
  const _Float16* p = base + (size_t)ln * rowstride + hl * 16;
  return cat8(*(const v8h*)p, *(const v8h*)(p + 8));
}

// B-matrix fragment (32x16 f16): lane&15 -> column N, lanes 0-15 K=0-15,
// lanes 16-31 K=16-31 (contiguous 16 halves per half-wave).
__device__ __forceinline__ v16h frag_b(const _Float16* base, int rowstride, int lane) {
  const int ln = lane & 15, hl = lane >> 4;
  const _Float16* p = base + (size_t)ln * rowstride + hl * 16;
  return cat8(*(const v8h*)p, *(const v8h*)(p + 8));
}

// CDNA5 async global->LDS DMA (ASYNCcnt-tracked). lds_off = wave-relative LDS
// byte address (low 32 bits of the generic pointer). The instruction offset is
// applied to BOTH the LDS and the global address, so chunk1 reuses the regs.
__device__ __forceinline__ void async_b128(unsigned int lds_off, const _Float16* g) {
  asm volatile("global_load_async_to_lds_b128 %0, %1, off"
               :: "v"(lds_off), "v"(g) : "memory");
}
__device__ __forceinline__ void async_b128_off16(unsigned int lds_off, const _Float16* g) {
  asm volatile("global_load_async_to_lds_b128 %0, %1, off offset:16"
               :: "v"(lds_off), "v"(g) : "memory");
}
__device__ __forceinline__ void wait_async0() {
  asm volatile("s_wait_asynccnt 0x0" ::: "memory");
}

// ---------------------------------------------------------------------------
// Elementwise helpers
// ---------------------------------------------------------------------------
__global__ void cast_f16_kernel(const float* __restrict__ in, _Float16* __restrict__ out, int n) {
  int i = blockIdx.x * 256 + threadIdx.x;
  if (i < n) out[i] = (_Float16)in[i];
}

// W[K][N] fp32 -> Wt[N][K] f16
__global__ void transpose_cast_kernel(const float* __restrict__ W, _Float16* __restrict__ Wt,
                                      int K, int N) {
  int i = blockIdx.x * 256 + threadIdx.x;
  if (i < K * N) {
    int kk = i / N, n = i - kk * N;
    Wt[(size_t)n * K + kk] = (_Float16)W[i];
  }
}

// v fp32 [B,T,H,DH] -> vt f16 [B,H,DH,T]
__global__ void transpose_v_kernel(const float* __restrict__ v, _Float16* __restrict__ vt, int T) {
  int i = blockIdx.x * 256 + threadIdx.x;   // over B*T*512
  int c  = i & 511;
  int bt = i >> 9;
  int t  = bt % T;
  int b  = bt / T;
  int hh = c >> 6, d = c & 63;
  vt[(((size_t)b * 8 + hh) * 64 + d) * (size_t)T + t] = (_Float16)v[i];
}

// RoPE: in fp32 [B,T,H*DH] -> out f16 (optionally pre-scaled by 1/sqrt(DH) for Q)
__global__ __launch_bounds__(256)
void rope_kernel(const float* __restrict__ in, _Float16* __restrict__ out, int T, float scale) {
  int bt = blockIdx.x;               // b*T + t
  int t  = bt % T;
  int i  = threadIdx.x;              // 256 threads = 8 heads * 32 pairs
  int hh = i >> 5, p = i & 31;
  float inv = __powf(10000.0f, -(float)p * (1.0f / 32.0f));
  float th  = (float)t * inv;
  float cs = cosf(th), sn = sinf(th);
  size_t base = (size_t)bt * 512 + hh * 64 + 2 * p;
  float e = in[base], o = in[base + 1];
  out[base]     = (_Float16)((e * cs - o * sn) * scale);
  out[base + 1] = (_Float16)((o * cs + e * sn) * scale);
}

// LayerNorm over rows of 512; writes f32 and/or f16
__global__ __launch_bounds__(256)
void ln_kernel(const float* __restrict__ x, const float* __restrict__ g,
               const float* __restrict__ bt, float* outf, _Float16* outh) {
  __shared__ float red[16];
  const int row = blockIdx.x, tid = threadIdx.x;
  const int lane = tid & 31, w = tid >> 5;
  const float* p = x + (size_t)row * 512;
  float a = p[tid], b = p[tid + 256];
  float s = a + b, q = a * a + b * b;
#pragma unroll
  for (int off = 16; off; off >>= 1) {
    s += __shfl_xor(s, off, 32);
    q += __shfl_xor(q, off, 32);
  }
  if (lane == 0) { red[w] = s; red[8 + w] = q; }
  __syncthreads();
  s = 0.0f; q = 0.0f;
#pragma unroll
  for (int i = 0; i < 8; ++i) { s += red[i]; q += red[8 + i]; }
  float mean = s * (1.0f / 512.0f);
  float var  = q * (1.0f / 512.0f) - mean * mean;
  float rstd = rsqrtf(var + 1e-5f);
  float y0 = (a - mean) * rstd * g[tid] + bt[tid];
  float y1 = (b - mean) * rstd * g[tid + 256] + bt[tid + 256];
  size_t o = (size_t)row * 512 + tid;
  if (outf) { outf[o] = y0; outf[o + 256] = y1; }
  if (outh) { outh[o] = (_Float16)y0; outh[o + 256] = (_Float16)y1; }
}

// ---------------------------------------------------------------------------
// WMMA GEMM: out[M][N] = act(alpha * A[M][K] @ Wt[N][K]^T + bias) + residual
// BM=BN=128, BK=32. 8 waves: 4 along M x 2 along N; wave tile 32x64 (2x4 WMMAs).
// Double-buffered LDS fed by global_load_async_to_lds_b128: tile i+1 DMAs
// while tile i computes; one barrier per K-iteration.
// LDS pitch 40 halves: rows 16B aligned, 20-word bank stride (conflict-free).
// ---------------------------------------------------------------------------
#define GPIT 40
__global__ __launch_bounds__(256)
void gemm_wmma(const _Float16* __restrict__ A, const _Float16* __restrict__ Wt,
               int M, int N, int K, float alpha,
               const float* __restrict__ bias, int relu,
               const float* residual, float* outf, _Float16* outh) {
  __shared__ __align__(16) _Float16 As[2][128 * GPIT];
  __shared__ __align__(16) _Float16 Bs[2][128 * GPIT];
  const int tid = threadIdx.x, lane = tid & 31, wave = tid >> 5;
  const int ln = lane & 15, hl = lane >> 4;
  const int m0 = blockIdx.y * 128, n0 = blockIdx.x * 128;
  const int wm = (wave >> 1) * 32;   // wave offset along M
  const int wn = (wave & 1) * 64;    // wave offset along N
  const int lr = tid >> 1;           // tile row this thread loads (0..127)
  const int lk = (tid & 1) * 16;     // k-offset (0 or 16 halves)

  const _Float16* gA = A  + (size_t)(m0 + lr) * K + lk;
  const _Float16* gB = Wt + (size_t)(n0 + lr) * K + lk;
  const unsigned int ldsA = (unsigned int)(size_t)(&As[0][lr * GPIT + lk]);
  const unsigned int ldsB = (unsigned int)(size_t)(&Bs[0][lr * GPIT + lk]);
  const unsigned int bufStride = 128 * GPIT * 2;  // bytes between buffers

  auto issue = [&](int buf, int k0) {
    unsigned int la = ldsA + (unsigned int)buf * bufStride;
    unsigned int lb = ldsB + (unsigned int)buf * bufStride;
    async_b128(la, gA + k0);           // A chunk halves [0,8)
    async_b128_off16(la, gA + k0);     // A chunk halves [8,16)
    async_b128(lb, gB + k0);
    async_b128_off16(lb, gB + k0);
  };

  v8f acc[2][4] = {};
  const int ntiles = K >> 5;

  issue(0, 0);
  wait_async0();
  __syncthreads();                     // tile 0 resident

  int buf = 0;
  for (int i = 0; i < ntiles; ++i) {
    if (i + 1 < ntiles) issue(buf ^ 1, (i + 1) << 5);  // DMA next tile

    const _Float16* as = As[buf];
    const _Float16* bs = Bs[buf];
    v16h af[2], bf[4];
#pragma unroll
    for (int mt = 0; mt < 2; ++mt) {
      const _Float16* pa = as + (wm + mt * 16 + ln) * GPIT + hl * 16;
      af[mt] = cat8(*(const v8h*)pa, *(const v8h*)(pa + 8));
    }
#pragma unroll
    for (int nt = 0; nt < 4; ++nt) {
      const _Float16* pb = bs + (wn + nt * 16 + ln) * GPIT + hl * 16;
      bf[nt] = cat8(*(const v8h*)pb, *(const v8h*)(pb + 8));
    }
#pragma unroll
    for (int mt = 0; mt < 2; ++mt)
#pragma unroll
      for (int nt = 0; nt < 4; ++nt)
        acc[mt][nt] = wmma_f16(af[mt], bf[nt], acc[mt][nt]);

    wait_async0();     // own DMAs for tile i+1 landed
    __syncthreads();   // all waves done reading buf + all DMAs visible
    buf ^= 1;
  }

  // Epilogue (C layout: VGPR r -> M = r + 8*halfwave, N = lane&15)
#pragma unroll
  for (int mt = 0; mt < 2; ++mt)
#pragma unroll
    for (int nt = 0; nt < 4; ++nt)
#pragma unroll
      for (int r = 0; r < 8; ++r) {
        int row = m0 + wm + mt * 16 + r + 8 * hl;
        int col = n0 + wn + nt * 16 + ln;
        float v = acc[mt][nt][r] * alpha;
        if (bias) v += bias[col];
        if (relu) v = fmaxf(v, 0.0f);
        size_t idx = (size_t)row * N + col;
        if (residual) v += residual[idx];
        if (outf) outf[idx] = v;
        if (outh) outh[idx] = (_Float16)v;
      }
}

// ---------------------------------------------------------------------------
// Windowed flash attention. One wave per 16-query tile; 8 waves/block.
// q: f16 [B,T,H,DH] (pre-scaled by 1/sqrt(DH)), k: f16 [B,T,H,DH],
// vt: f16 [B,H,DH,T], out: f16 [B,T,H*DH].
// ---------------------------------------------------------------------------
#define APIT 40
__global__ __launch_bounds__(256)
void attn_wmma(const _Float16* __restrict__ q, const _Float16* __restrict__ k,
               const _Float16* __restrict__ vt, _Float16* __restrict__ out,
               int T, int half) {
  __shared__ __align__(16) _Float16 Ps[8 * 16 * APIT];   // per-wave 16x32 P tile
  const int lane = threadIdx.x & 31, wave = threadIdx.x >> 5;
  const int ln = lane & 15, hl = lane >> 4;
  const int nqb = T >> 7;                  // query blocks of 128 per (b,h)
  const int bh = blockIdx.x / nqb, qb = blockIdx.x % nqb;
  const int b = bh >> 3, h = bh & 7;
  const int qt0 = qb * 128 + wave * 16;

  const _Float16* qbase = q + (((size_t)b * T + qt0) * 8 + h) * 64;
  const _Float16* kbase = k + ((size_t)b * T * 8 + h) * 64;
  const _Float16* vbase = vt + ((size_t)(b * 8 + h)) * 64 * (size_t)T;
  _Float16* ps = Ps + wave * 16 * APIT;

  // Q fragments over dh (K-dim split 0-31 / 32-63), cached for whole loop
  v16h qa0 = frag_a(qbase, 512, lane);
  v16h qa1 = frag_a(qbase + 32, 512, lane);

  v8f o[4] = {};
  float m[8], l[8];
#pragma unroll
  for (int r = 0; r < 8; ++r) { m[r] = -1e30f; l[r] = 0.0f; }

  int lo = qt0 - half; if (lo < 0) lo = 0; lo &= ~31;
  int hi = qt0 + 15 + half + 1; if (hi > T) hi = T;

  for (int kb = lo; kb < hi; kb += 32) {
    // ---- scores: two 16x16 tiles (keys kb..kb+15, kb+16..kb+31) ----
    v8f s0 = {}, s1 = {};
    s0 = wmma_f16(qa0, frag_b(kbase + (size_t)kb * 512,        512, lane), s0);
    s0 = wmma_f16(qa1, frag_b(kbase + (size_t)kb * 512 + 32,   512, lane), s0);
    s1 = wmma_f16(qa0, frag_b(kbase + (size_t)(kb + 16) * 512,      512, lane), s1);
    s1 = wmma_f16(qa1, frag_b(kbase + (size_t)(kb + 16) * 512 + 32, 512, lane), s1);

    // ---- mask + online softmax (row stats live in one 16-lane half) ----
    float sc[8];
#pragma unroll
    for (int r = 0; r < 8; ++r) {
      int row = qt0 + r + 8 * hl;
      int c0 = kb + ln, c1 = c0 + 16;
      float x0 = s0[r], x1 = s1[r];
      int d0 = row - c0, d1 = row - c1;
      if (d0 < -half || d0 > half) x0 = -1e30f;
      if (d1 < -half || d1 > half) x1 = -1e30f;
      float mr = fmaxf(x0, x1);
#pragma unroll
      for (int off = 1; off < 16; off <<= 1) mr = fmaxf(mr, __shfl_xor(mr, off, 32));
      float mn = fmaxf(m[r], mr);
      float scale = __expf(m[r] - mn);
      float p0 = __expf(x0 - mn), p1 = __expf(x1 - mn);
      float ss = p0 + p1;
#pragma unroll
      for (int off = 1; off < 16; off <<= 1) ss += __shfl_xor(ss, off, 32);
      l[r] = l[r] * scale + ss;
      m[r] = mn;
      sc[r] = scale;
      int prow = r + 8 * hl;
      ps[prow * APIT + ln]      = (_Float16)p0;
      ps[prow * APIT + ln + 16] = (_Float16)p1;
    }
#pragma unroll
    for (int d = 0; d < 4; ++d)
#pragma unroll
      for (int r = 0; r < 8; ++r) o[d][r] *= sc[r];

    // ---- O += P @ V (P reread in A layout from wave-private LDS;
    //      DS ops are in-order within a wave, no barrier needed) ----
    v16h pa = frag_a(ps, APIT, lane);
#pragma unroll
    for (int d = 0; d < 4; ++d)
      o[d] = wmma_f16(pa, frag_b(vbase + (size_t)(d * 16) * T + kb, T, lane), o[d]);
  }

  // ---- normalize and write f16 [B,T,H*DH] ----
#pragma unroll
  for (int r = 0; r < 8; ++r) {
    int row = qt0 + r + 8 * hl;
    float inv = 1.0f / l[r];
    size_t base = ((size_t)b * T + row) * 512 + h * 64 + ln;
#pragma unroll
    for (int d = 0; d < 4; ++d) out[base + d * 16] = (_Float16)(o[d][r] * inv);
  }
}

// ---------------------------------------------------------------------------
// Host-side orchestration
// ---------------------------------------------------------------------------
static inline char* bump(char*& p, size_t bytes) {
  char* r = p;
  p += (bytes + 255) & ~(size_t)255;
  return r;
}

extern "C" void kernel_launch(void* const* d_in, const int* in_sizes, int n_in,
                              void* d_out, int out_size, void* d_ws, size_t ws_size,
                              hipStream_t stream) {
  (void)in_sizes; (void)n_in; (void)out_size; (void)ws_size;
  const int Bt = 2, T = 2048, F = 128, E = 512, MT = Bt * T, FF = 4 * E;

  const float* X     = (const float*)d_in[0];
  const float* EMBW  = (const float*)d_in[1];
  const float* LNF_S = (const float*)d_in[2];
  const float* LNF_B = (const float*)d_in[3];

  // ---- workspace layout ----
  char* p = (char*)d_ws;
  _Float16* xh   = (_Float16*)bump(p, (size_t)MT * F * 2);
  _Float16* embt = (_Float16*)bump(p, (size_t)E * F * 2);
  _Float16* wqt  = (_Float16*)bump(p, (size_t)E * E * 2);
  _Float16* wkt  = (_Float16*)bump(p, (size_t)E * E * 2);
  _Float16* wvt  = (_Float16*)bump(p, (size_t)E * E * 2);
  _Float16* wot  = (_Float16*)bump(p, (size_t)E * E * 2);
  _Float16* w1t  = (_Float16*)bump(p, (size_t)FF * E * 2);
  _Float16* w2t  = (_Float16*)bump(p, (size_t)E * FF * 2);
  float*    h    = (float*)   bump(p, (size_t)MT * E * 4);
  _Float16* hn   = (_Float16*)bump(p, (size_t)MT * E * 2);
  float*    qf   = (float*)   bump(p, (size_t)MT * E * 4);
  float*    kf   = (float*)   bump(p, (size_t)MT * E * 4);   // adjacent to qf
  float*    vf   = (float*)   bump(p, (size_t)MT * E * 4);
  _Float16* qh   = (_Float16*)bump(p, (size_t)MT * E * 2);
  _Float16* kh   = (_Float16*)bump(p, (size_t)MT * E * 2);
  _Float16* vth  = (_Float16*)bump(p, (size_t)MT * E * 2);
  _Float16* aoh  = (_Float16*)bump(p, (size_t)MT * E * 2);
  _Float16* f1h  = (_Float16*)qf;  // 16MB alias spanning qf+kf (free at FFN time)

  dim3 blk(256);
  auto gemm = [&](const _Float16* A, const _Float16* Wt, int M, int N, int K, float alpha,
                  const float* bias, int relu, const float* res, float* of, _Float16* oh) {
    dim3 grid(N / 128, M / 128);
    gemm_wmma<<<grid, blk, 0, stream>>>(A, Wt, M, N, K, alpha, bias, relu, res, of, oh);
  };

  // h = (x @ emb) * sqrt(EMB)
  cast_f16_kernel<<<(MT * F + 255) / 256, blk, 0, stream>>>(X, xh, MT * F);
  transpose_cast_kernel<<<(F * E + 255) / 256, blk, 0, stream>>>(EMBW, embt, F, E);
  gemm(xh, embt, MT, E, F, sqrtf((float)E), nullptr, 0, nullptr, h, nullptr);

  for (int l = 0; l < 2; ++l) {
    const float* wq  = (const float*)d_in[4 + 12 * l + 0];
    const float* wk  = (const float*)d_in[4 + 12 * l + 1];
    const float* wv  = (const float*)d_in[4 + 12 * l + 2];
    const float* wo  = (const float*)d_in[4 + 12 * l + 3];
    const float* w1  = (const float*)d_in[4 + 12 * l + 4];
    const float* b1  = (const float*)d_in[4 + 12 * l + 5];
    const float* w2  = (const float*)d_in[4 + 12 * l + 6];
    const float* b2  = (const float*)d_in[4 + 12 * l + 7];
    const float* l1s = (const float*)d_in[4 + 12 * l + 8];
    const float* l1b = (const float*)d_in[4 + 12 * l + 9];
    const float* l2s = (const float*)d_in[4 + 12 * l + 10];
    const float* l2b = (const float*)d_in[4 + 12 * l + 11];

    transpose_cast_kernel<<<(E * E + 255) / 256, blk, 0, stream>>>(wq, wqt, E, E);
    transpose_cast_kernel<<<(E * E + 255) / 256, blk, 0, stream>>>(wk, wkt, E, E);
    transpose_cast_kernel<<<(E * E + 255) / 256, blk, 0, stream>>>(wv, wvt, E, E);
    transpose_cast_kernel<<<(E * E + 255) / 256, blk, 0, stream>>>(wo, wot, E, E);
    transpose_cast_kernel<<<(E * FF + 255) / 256, blk, 0, stream>>>(w1, w1t, E, FF);
    transpose_cast_kernel<<<(FF * E + 255) / 256, blk, 0, stream>>>(w2, w2t, FF, E);

    // attention
    ln_kernel<<<MT, blk, 0, stream>>>(h, l1s, l1b, nullptr, hn);
    gemm(hn, wqt, MT, E, E, 1.0f, nullptr, 0, nullptr, qf, nullptr);
    gemm(hn, wkt, MT, E, E, 1.0f, nullptr, 0, nullptr, kf, nullptr);
    gemm(hn, wvt, MT, E, E, 1.0f, nullptr, 0, nullptr, vf, nullptr);
    rope_kernel<<<MT, blk, 0, stream>>>(qf, qh, T, 0.125f);   // fold 1/sqrt(64)
    rope_kernel<<<MT, blk, 0, stream>>>(kf, kh, T, 1.0f);
    transpose_v_kernel<<<(MT * E) / 256, blk, 0, stream>>>(vf, vth, T);
    attn_wmma<<<Bt * 8 * (T / 128), blk, 0, stream>>>(qh, kh, vth, aoh, T, 256);
    gemm(aoh, wot, MT, E, E, 1.0f, nullptr, 0, h, h, nullptr);  // h += attn @ wo

    // FFN
    ln_kernel<<<MT, blk, 0, stream>>>(h, l2s, l2b, nullptr, hn);
    gemm(hn, w1t, MT, FF, E, 1.0f, b1, 1, nullptr, nullptr, f1h);
    gemm(f1h, w2t, MT, E, FF, 1.0f, b2, 0, h, h, nullptr);      // h += ffn
  }

  ln_kernel<<<MT, blk, 0, stream>>>(h, LNF_S, LNF_B, (float*)d_out, nullptr);
}
